// FigLinear_54820962566631
// MI455X (gfx1250) — compile-verified, matrix operand order checked by reading, not run on hardware
//
#include <hip/hip_runtime.h>
#include <stdint.h>

// Problem constants (from the reference)
#define IN_F   4096
#define OUT_F  4096
#define LORA_R 16
#define M_TOT  8192           // B=4 * S=2048

constexpr int BM = 128, BN = 128, BK = 32;
constexpr int NSTAGE  = IN_F / BK;    // 128 K-stages
constexpr int LDSTR_E = 40;           // bf16 elems per LDS row (32 + 8 pad -> 80B stride, conflict-free)
constexpr int LDSTR_W = LDSTR_E / 2;  // u32 per LDS row (20)

typedef __bf16  bf16_t;
typedef bf16_t  v16bf __attribute__((ext_vector_type(16)));
typedef float   v8f   __attribute__((ext_vector_type(8)));
typedef float   v2f   __attribute__((ext_vector_type(2)));

union FragU { uint4 u[2]; v16bf v; };

// Split workgroup barrier (gfx1250): signal early, overlap work, wait late.
// s_wait_dscnt 0 before signal guarantees (a) our LDS tile stores are visible to
// the workgroup and (b) our previous-stage fragment ds_loads have completed (WAR).
__device__ __forceinline__ void wg_barrier_signal_fenced() {
  asm volatile("s_wait_dscnt 0x0" ::: "memory");
  asm volatile("s_barrier_signal -1" ::: "memory");
}
__device__ __forceinline__ void wg_barrier_wait() {
  asm volatile("s_barrier_wait -1" ::: "memory");
}

// round-half-up f32x2 -> packed bf16x2 via V_PERM_B32 (a -> low half, b -> high half)
__device__ __forceinline__ uint32_t pack_bf16(float a, float b) {
  uint32_t ua = __builtin_bit_cast(uint32_t, a) + 0x8000u;
  uint32_t ub = __builtin_bit_cast(uint32_t, b) + 0x8000u;
  // result byte0,1 = ua bytes 2,3 ; byte2,3 = ub bytes 2,3
  return __builtin_amdgcn_perm(ub, ua, 0x07060302u);
}

// ---------------- Pass 1: t = x @ lora_A  (M_TOT x 16) ----------------
__global__ __launch_bounds__(256) void lora_xA_kernel(const float* __restrict__ x,
                                                      const float* __restrict__ A,
                                                      float* __restrict__ t) {
  __shared__ float red[256][17];   // +1 pad vs bank conflicts
  const int tid = threadIdx.x;
  const int m   = blockIdx.x;

  const float4* xr = (const float4*)(x + (size_t)m * IN_F + tid * 16);
  float4 xv[4];
#pragma unroll
  for (int i = 0; i < 4; ++i) xv[i] = xr[i];

  float p[16];
#pragma unroll
  for (int r = 0; r < 16; ++r) p[r] = 0.f;

#pragma unroll
  for (int i = 0; i < 4; ++i) {
    const float xk[4] = { xv[i].x, xv[i].y, xv[i].z, xv[i].w };
#pragma unroll
    for (int j = 0; j < 4; ++j) {
      const int k = tid * 16 + i * 4 + j;
      const float4* ar = (const float4*)(A + (size_t)k * LORA_R);
      const float4 a0 = ar[0], a1 = ar[1], a2 = ar[2], a3 = ar[3];
      p[0]  += xk[j] * a0.x;  p[1]  += xk[j] * a0.y;  p[2]  += xk[j] * a0.z;  p[3]  += xk[j] * a0.w;
      p[4]  += xk[j] * a1.x;  p[5]  += xk[j] * a1.y;  p[6]  += xk[j] * a1.z;  p[7]  += xk[j] * a1.w;
      p[8]  += xk[j] * a2.x;  p[9]  += xk[j] * a2.y;  p[10] += xk[j] * a2.z;  p[11] += xk[j] * a2.w;
      p[12] += xk[j] * a3.x;  p[13] += xk[j] * a3.y;  p[14] += xk[j] * a3.z;  p[15] += xk[j] * a3.w;
    }
  }
#pragma unroll
  for (int r = 0; r < 16; ++r) red[tid][r] = p[r];
  __syncthreads();
#pragma unroll 1
  for (int off = 128; off > 0; off >>= 1) {
    if (tid < off) {
#pragma unroll
      for (int r = 0; r < 16; ++r) red[tid][r] += red[tid + off][r];
    }
    __syncthreads();
  }
  if (tid < 16) t[(size_t)m * LORA_R + tid] = red[0][tid];
}

// ---------------- Pass 2 helpers ----------------
__device__ __forceinline__ void stage_gload(const float4* gx, const int4* gq, const float* gsc, int s,
                                            float4 (&xr)[4], int4 (&qr)[4], float& sc) {
  const float4* px = gx + (size_t)s * (BK / 4);
  const int4*   pq = gq + (size_t)s * (BK / 4);
#pragma unroll
  for (int i = 0; i < 4; ++i) { xr[i] = px[i]; qr[i] = pq[i]; }
  sc = gsc[s >> 2];   // group of 128: k-run of 32 never crosses a group boundary
}

// Dequant one index pair through the 256-entry float2 pair-LUT:
// one ds_load_b64 + one v_pk_mul_f32 + one v_perm pack for two elements.
__device__ __forceinline__ uint32_t dq_pair(const v2f* lut, int qa, int qb, v2f scv) {
  const v2f w = lut[qa + (qb << 4)] * scv;
  return pack_bf16(w[0], w[1]);
}

__device__ __forceinline__ void stage_sstore(uint32_t* Xb, uint32_t* Wb, const v2f* lut,
                                             const float4 (&xr)[4], const int4 (&qr)[4], float sc,
                                             int row, int kh) {
  // x tile -> bf16
  uint4 v0, v1;
  v0.x = pack_bf16(xr[0].x, xr[0].y);  v0.y = pack_bf16(xr[0].z, xr[0].w);
  v0.z = pack_bf16(xr[1].x, xr[1].y);  v0.w = pack_bf16(xr[1].z, xr[1].w);
  v1.x = pack_bf16(xr[2].x, xr[2].y);  v1.y = pack_bf16(xr[2].z, xr[2].w);
  v1.z = pack_bf16(xr[3].x, xr[3].y);  v1.w = pack_bf16(xr[3].z, xr[3].w);
  uint32_t* dx = Xb + row * LDSTR_W + (kh >> 1);
  ((uint4*)dx)[0]       = v0;
  ((uint4*)(dx + 4))[0] = v1;

  // W tile: pair-LUT gather * packed scale -> bf16
  const v2f scv = { sc, sc };
  uint4 w0, w1;
  w0.x = dq_pair(lut, qr[0].x, qr[0].y, scv);
  w0.y = dq_pair(lut, qr[0].z, qr[0].w, scv);
  w0.z = dq_pair(lut, qr[1].x, qr[1].y, scv);
  w0.w = dq_pair(lut, qr[1].z, qr[1].w, scv);
  w1.x = dq_pair(lut, qr[2].x, qr[2].y, scv);
  w1.y = dq_pair(lut, qr[2].z, qr[2].w, scv);
  w1.z = dq_pair(lut, qr[3].x, qr[3].y, scv);
  w1.w = dq_pair(lut, qr[3].z, qr[3].w, scv);
  uint32_t* dw = Wb + row * LDSTR_W + (kh >> 1);
  ((uint4*)dw)[0]       = w0;
  ((uint4*)(dw + 4))[0] = w1;
}

// ISA 7.12.2 16-bit A/B layout: lane l holds row (l&15); K runs [(l>>4)*8,+8) and +16.
__device__ __forceinline__ v16bf frag_ld(const uint32_t* buf, int r, int lane) {
  const uint32_t* p = buf + r * LDSTR_W + ((lane >> 4) << 2);
  FragU f;
  f.u[0] = ((const uint4*)p)[0];        // K run 0 (8 bf16)
  f.u[1] = ((const uint4*)(p + 8))[0];  // K run +16 (8 bf16)
  return f.v;
}

__device__ __forceinline__ void do_wmma(const uint32_t* Xb, const uint32_t* Wb,
                                        int wm, int wn, int lane, v8f (&acc)[4][2]) {
  const int rl = lane & 15;
  v16bf af[4];
#pragma unroll
  for (int mt = 0; mt < 4; ++mt) af[mt] = frag_ld(Xb, wm * 64 + mt * 16 + rl, lane);
#pragma unroll
  for (int nt = 0; nt < 2; ++nt) {
    const v16bf bf = frag_ld(Wb, wn * 32 + nt * 16 + rl, lane);
#pragma unroll
    for (int mt = 0; mt < 4; ++mt)
      acc[mt][nt] = __builtin_amdgcn_wmma_f32_16x16x32_bf16(
          false, af[mt], false, bf, (short)0, acc[mt][nt], false, false);
  }
}

// ---------------- Pass 2: out = dequant-GEMM + 2*(t @ lora_B) + bias ----------------
__global__ __launch_bounds__(256) void figlinear_main(
    const float* __restrict__ x, const int* __restrict__ qidx,
    const float* __restrict__ cbg, const float* __restrict__ scales,
    const float* __restrict__ loraB, const float* __restrict__ bias,
    const float* __restrict__ tbuf, float* __restrict__ out) {
  __shared__ __align__(16) uint32_t Xl[2][BM * LDSTR_W];   // 2 x 10KB
  __shared__ __align__(16) uint32_t Wl[2][BM * LDSTR_W];   // 2 x 10KB
  __shared__ __align__(8)  v2f      lut[256];              // pair-LUT: (cb[i&15], cb[i>>4])

  const int tid  = threadIdx.x;
  const int lane = tid & 31;
  const int wid  = tid >> 5;
  const int wm   = wid >> 2;     // 0..1  -> M offset wm*64
  const int wn   = wid & 3;      // 0..3  -> N offset wn*32
  const int n0   = blockIdx.x * BN;
  const int m0   = blockIdx.y * BM;

  {
    const v2f e = { cbg[tid & 15], cbg[tid >> 4] };
    lut[tid] = e;
  }

  const int row = tid >> 1;           // 0..127
  const int kh  = (tid & 1) * 16;     // 0 or 16 within the 32-wide K stage
  const float4* gx  = (const float4*)(x + (size_t)(m0 + row) * IN_F + kh);
  const int4*   gq  = (const int4*)(qidx + (size_t)(n0 + row) * IN_F + kh);
  const float*  gsc = scales + (size_t)(n0 + row) * (IN_F / 128);

  v8f acc[4][2];
#pragma unroll
  for (int mt = 0; mt < 4; ++mt)
#pragma unroll
    for (int nt = 0; nt < 2; ++nt)
#pragma unroll
      for (int j = 0; j < 8; ++j) acc[mt][nt][j] = 0.f;

  float4 xrA[4], xrB[4];
  int4   qrA[4], qrB[4];
  float  scA = 0.f, scB = 0.f;

  stage_gload(gx, gq, gsc, 0, xrA, qrA, scA);
  __syncthreads();   // LUT visible before first dequant

  // Double-buffered pipeline per stage:
  //   ds_store(tile s) -> wait_dscnt 0 + barrier_signal -> issue global loads(s+1)
  //   -> barrier_wait -> WMMA(tile s)
  // The global prefetch is issued inside the split barrier so it overlaps both the
  // other waves' barrier arrival and the WMMA phase, and the pre-signal wait only
  // drains DScnt (no loadcnt serialization).
#pragma unroll 1
  for (int s = 0; s < NSTAGE; s += 2) {
    stage_sstore(Xl[0], Wl[0], lut, xrA, qrA, scA, row, kh);
    wg_barrier_signal_fenced();
    if (s + 1 < NSTAGE) stage_gload(gx, gq, gsc, s + 1, xrB, qrB, scB);
    wg_barrier_wait();
    do_wmma(Xl[0], Wl[0], wm, wn, lane, acc);

    stage_sstore(Xl[1], Wl[1], lut, xrB, qrB, scB, row, kh);
    wg_barrier_signal_fenced();
    if (s + 2 < NSTAGE) stage_gload(gx, gq, gsc, s + 2, xrA, qrA, scA);
    wg_barrier_wait();
    do_wmma(Xl[1], Wl[1], wm, wn, lane, acc);
  }

  // Epilogue: C/D layout -> element j of v8f is (M = j + 8*(lane>>4), N = lane&15)
  const int lh = lane >> 4;
  const int ll = lane & 15;

  float bcol[2][16];
  float bv[2];
  int   gn[2];
#pragma unroll
  for (int nt = 0; nt < 2; ++nt) {
    gn[nt] = n0 + wn * 32 + nt * 16 + ll;
#pragma unroll
    for (int r = 0; r < 16; ++r) bcol[nt][r] = loraB[(size_t)r * OUT_F + gn[nt]];
    bv[nt] = bias[gn[nt]];
  }

#pragma unroll
  for (int mt = 0; mt < 4; ++mt) {
#pragma unroll
    for (int j = 0; j < 8; ++j) {
      const int gm = m0 + wm * 64 + mt * 16 + lh * 8 + j;
      const float4* tr = (const float4*)(tbuf + (size_t)gm * LORA_R);
      const float4 t0 = tr[0], t1 = tr[1], t2 = tr[2], t3 = tr[3];
#pragma unroll
      for (int nt = 0; nt < 2; ++nt) {
        const float l =
            t0.x * bcol[nt][0]  + t0.y * bcol[nt][1]  + t0.z * bcol[nt][2]  + t0.w * bcol[nt][3]  +
            t1.x * bcol[nt][4]  + t1.y * bcol[nt][5]  + t1.z * bcol[nt][6]  + t1.w * bcol[nt][7]  +
            t2.x * bcol[nt][8]  + t2.y * bcol[nt][9]  + t2.z * bcol[nt][10] + t2.w * bcol[nt][11] +
            t3.x * bcol[nt][12] + t3.y * bcol[nt][13] + t3.z * bcol[nt][14] + t3.w * bcol[nt][15];
        out[(size_t)gm * OUT_F + gn[nt]] = acc[mt][nt][j] + 2.0f * l + bv[nt];
      }
    }
  }
}

// ---------------- Launcher ----------------
extern "C" void kernel_launch(void* const* d_in, const int* in_sizes, int n_in,
                              void* d_out, int out_size, void* d_ws, size_t ws_size,
                              hipStream_t stream) {
  (void)in_sizes; (void)n_in; (void)out_size; (void)ws_size;
  const float* x      = (const float*)d_in[0];
  const int*   qidx   = (const int*)d_in[1];
  const float* cb     = (const float*)d_in[2];
  const float* sc     = (const float*)d_in[3];
  const float* loraA  = (const float*)d_in[4];
  const float* loraB  = (const float*)d_in[5];
  const float* bias   = (const float*)d_in[6];
  float* out  = (float*)d_out;
  float* tbuf = (float*)d_ws;   // needs M_TOT * 16 * 4 = 512 KB of workspace

  lora_xA_kernel<<<M_TOT, 256, 0, stream>>>(x, loraA, tbuf);

  dim3 grid(OUT_F / BN, M_TOT / BM);  // 32 x 64
  figlinear_main<<<grid, 256, 0, stream>>>(x, qidx, cb, sc, loraB, bias, tbuf, out);
}
